// FusedMoEModularKernel_46909632807489
// MI455X (gfx1250) — matrix operand order repaired and checked by reading, not compiled, not run on hardware
//
#include <hip/hip_runtime.h>

// ---------------------------------------------------------------------------
// FusedMoE for MI455X (gfx1250), grouped-GEMM formulation, atomic-free.
//   slot s=(m,t): expert e=topk_ids[m][t], weight w=tw[m][t]
//   out[m] = sum_t tw[m][t] * ( silu(x_m@W1g_e)*(x_m@W1u_e) ) @ W2_e
// Rows gathered per expert (padded to 128-row tiles): O(M*TOPK) GEMM rows
// instead of O(M*E) (~3.5x fewer FLOPs). bf16 WMMA datapath, fp32 accumulate.
// Combine stage is a streaming 2-way weighted reduce (no atomics).
// ---------------------------------------------------------------------------

typedef __bf16 bf16_t;
typedef __attribute__((ext_vector_type(16))) __bf16 v16bf;
typedef __attribute__((ext_vector_type(8)))  __bf16 v8bf;
typedef __attribute__((ext_vector_type(8)))  float  v8f;

#define MOE_M    4096
#define MOE_K    1024
#define MOE_E    8
#define MOE_N    1024
#define MOE_TOPK 2
#define NSLOT    (MOE_M * MOE_TOPK)   // 8192 assignment slots
#define MAXTILES 72                   // >= sum_e ceil(cnt_e/128) worst case (71)
#define PADROWS  (MAXTILES * 128)     // 9216 padded gathered rows

union FragU { v16bf f; v8bf h[2]; };

// A-fragment, 16x32 bf16, row-major source, leading dim ld (elements).
__device__ __forceinline__ v16bf load_frag_A(const bf16_t* __restrict__ p,
                                             int row, int kbase, int ld, int lane) {
  int r  = row + (lane & 15);
  int k0 = kbase + ((lane >> 4) << 4);           // 0 or 16
  const bf16_t* q = p + (long)r * ld + k0;
  FragU u;
  u.h[0] = *(const v8bf*)(q);                    // K = k0 .. k0+7
  u.h[1] = *(const v8bf*)(q + 8);                // K = k0+8 .. k0+15
  return u.f;
}

// B-fragment, 32x16 bf16, from pre-transposed weight (output column contiguous).
__device__ __forceinline__ v16bf load_frag_B(const bf16_t* __restrict__ p,
                                             int col, int kbase, int ld, int lane) {
  int c  = col + (lane & 15);
  int k0 = kbase + ((lane >> 4) << 4);           // 0 or 16
  const bf16_t* q = p + (long)c * ld + k0;
  FragU u;
  u.h[0] = *(const v8bf*)(q);
  u.h[1] = *(const v8bf*)(q + 8);
  return u.f;
}

__device__ __forceinline__ v8f wmma_bf16(v16bf a, v16bf b, v8f c) {
  return __builtin_amdgcn_wmma_f32_16x16x32_bf16(
      false, a, false, b, (short)0, c, false, false);
}

// --------------------------- small prep kernels ----------------------------

// in: [E][R][C] fp32 -> out: [E][C][R] bf16 (writes coalesced; src L2-resident)
__global__ void transpose_cvt_bf16(const float* __restrict__ in,
                                   bf16_t* __restrict__ out,
                                   int E, int R, int C) {
  long total  = (long)E * R * C;
  long stride = (long)gridDim.x * blockDim.x;
  for (long i = (long)blockIdx.x * blockDim.x + threadIdx.x; i < total; i += stride) {
    long rc  = (long)R * C;
    long e   = i / rc;
    long rem = i - e * rc;
    long c   = rem / R;
    long r   = rem - c * R;
    out[i] = (bf16_t)in[e * rc + r * C + c];
  }
}

// Deterministic expert-sorted permutation + inverse map + tile table.
// One block, 256 threads. perm[pos] = slot id or -1 (pad); inv[slot] = pos.
__global__ void build_perm(const long long* __restrict__ ids,  // [M][TOPK] int64
                           int* __restrict__ perm,             // [PADROWS]
                           int* __restrict__ inv,              // [NSLOT]
                           int* __restrict__ tile_e,           // [MAXTILES]
                           int* __restrict__ tile_m) {         // [MAXTILES]
  __shared__ int part[256][MOE_E];
  const int t   = threadIdx.x;
  const int per = NSLOT / 256;                   // 32 slots per thread

  for (int i = t; i < PADROWS; i += 256) perm[i] = -1;

  int lc[MOE_E];
  for (int e = 0; e < MOE_E; ++e) lc[e] = 0;
  for (int i = 0; i < per; ++i) lc[(int)ids[t * per + i]]++;
  for (int e = 0; e < MOE_E; ++e) part[t][e] = lc[e];
  __syncthreads();

  if (t == 0) {
    int pcnt[MOE_E], off[MOE_E];
    for (int e = 0; e < MOE_E; ++e) {
      int s = 0;
      for (int j = 0; j < 256; ++j) s += part[j][e];
      pcnt[e] = (s + 127) & ~127;                // pad each expert to 128 rows
    }
    int run = 0;
    for (int e = 0; e < MOE_E; ++e) { off[e] = run; run += pcnt[e]; }
    int cur[MOE_E];
    for (int e = 0; e < MOE_E; ++e) cur[e] = off[e];
    for (int j = 0; j < 256; ++j)                // serial exclusive scan (2048 adds)
      for (int e = 0; e < MOE_E; ++e) { int c = part[j][e]; part[j][e] = cur[e]; cur[e] += c; }
    int tid = 0;
    for (int e = 0; e < MOE_E; ++e)
      for (int mt = 0; mt < pcnt[e] / 128; ++mt) {
        tile_e[tid] = e; tile_m[tid] = off[e] + mt * 128; ++tid;
      }
    for (; tid < MAXTILES; ++tid) { tile_e[tid] = -1; tile_m[tid] = 0; }
  }
  __syncthreads();

  int pos[MOE_E];
  for (int e = 0; e < MOE_E; ++e) pos[e] = part[t][e];
  for (int i = 0; i < per; ++i) {
    int s = t * per + i;
    int e = (int)ids[s];
    int p = pos[e]++;
    perm[p] = s;
    inv[s]  = p;
  }
}

// Gather + fp32->bf16: Xg[pos] = bf16(X[row(perm[pos])]). Pads copy row 0
// (pad Yg rows are never referenced by inv, so no zeroing needed anywhere).
__global__ void gather_cvt_x(const float* __restrict__ X,
                             const int* __restrict__ perm,
                             bf16_t* __restrict__ Xg, int Kdim) {
  int pos = blockIdx.x;
  int s   = perm[pos];
  int m   = (s >= 0) ? (s >> 1) : 0;             // TOPK == 2
  const float* src = X  + (long)m   * Kdim;
  bf16_t*      dst = Xg + (long)pos * Kdim;
  for (int k = threadIdx.x * 4; k < Kdim; k += blockDim.x * 4) {
    float4 v = *(const float4*)(src + k);
    dst[k + 0] = (bf16_t)v.x; dst[k + 1] = (bf16_t)v.y;
    dst[k + 2] = (bf16_t)v.z; dst[k + 3] = (bf16_t)v.w;
  }
}

// ------------------------------ grouped GEMM 1 -----------------------------
// Hg[r][n] = silu(Xg@W1g_e)[r][n] * (Xg@W1u_e)[r][n] for this tile's expert.
// Block: 128 gathered rows x 64 gate cols (+ matching up cols); 8 waves,
// each wave owns a 32x32 gate tile AND the matching up tile: 8 WMMA / k-step.
__global__ __launch_bounds__(256)
void moe_gemm1_silu(const bf16_t* __restrict__ Xg,    // [PADROWS][K]
                    const bf16_t* __restrict__ W1T,   // [E][2N][K]
                    bf16_t* __restrict__ Hg,          // [PADROWS][N]
                    const int* __restrict__ tile_e,
                    const int* __restrict__ tile_m,
                    int Kdim, int Ndim) {
  const int e = tile_e[blockIdx.x];
  if (e < 0) return;                               // unused worst-case tile
  const int row0 = tile_m[blockIdx.x];

  const int lane = threadIdx.x & 31;
  const int wid  = threadIdx.x >> 5;
  const int wm   = wid & 3;
  const int wn   = wid >> 2;
  const int m0   = row0 + wm * 32;
  const int n0   = blockIdx.y * 64 + wn * 32;
  const bf16_t* Wg = W1T + (size_t)e * 2 * Ndim * Kdim;
  const bf16_t* Wu = Wg + (long)Ndim * Kdim;

  v8f ag[2][2], au[2][2];
  for (int i = 0; i < 2; ++i)
    for (int j = 0; j < 2; ++j) { ag[i][j] = (v8f)(0.0f); au[i][j] = (v8f)(0.0f); }

  for (int k = 0; k < Kdim; k += 32) {
    v16bf a0  = load_frag_A(Xg, m0,      k, Kdim, lane);
    v16bf a1  = load_frag_A(Xg, m0 + 16, k, Kdim, lane);
    v16bf bg0 = load_frag_B(Wg, n0,      k, Kdim, lane);
    v16bf bg1 = load_frag_B(Wg, n0 + 16, k, Kdim, lane);
    v16bf bu0 = load_frag_B(Wu, n0,      k, Kdim, lane);
    v16bf bu1 = load_frag_B(Wu, n0 + 16, k, Kdim, lane);
    ag[0][0] = wmma_bf16(a0, bg0, ag[0][0]);
    ag[0][1] = wmma_bf16(a0, bg1, ag[0][1]);
    ag[1][0] = wmma_bf16(a1, bg0, ag[1][0]);
    ag[1][1] = wmma_bf16(a1, bg1, ag[1][1]);
    au[0][0] = wmma_bf16(a0, bu0, au[0][0]);
    au[0][1] = wmma_bf16(a0, bu1, au[0][1]);
    au[1][0] = wmma_bf16(a1, bu0, au[1][0]);
    au[1][1] = wmma_bf16(a1, bu1, au[1][1]);
  }

  const int nl = lane & 15;
  const int ml = (lane >> 4) << 3;
  for (int mi = 0; mi < 2; ++mi)
    for (int ni = 0; ni < 2; ++ni) {
      int mbase = m0 + mi * 16 + ml;
      int n     = n0 + ni * 16 + nl;
      for (int r = 0; r < 8; ++r) {
        float g = ag[mi][ni][r];
        float u = au[mi][ni][r];
        float s = g / (1.0f + __expf(-g));       // silu(g)
        Hg[(long)(mbase + r) * Ndim + n] = (bf16_t)(s * u);
      }
    }
}

// ------------------------------ grouped GEMM 2 -----------------------------
// Yg[row][j] = (Hg @ W2_e)[row][j]  (unscaled, plain streaming stores).
__global__ __launch_bounds__(256)
void moe_gemm2(const bf16_t* __restrict__ Hg,    // [PADROWS][N]
               const bf16_t* __restrict__ W2T,   // [E][K][N]
               const int* __restrict__ tile_e,
               const int* __restrict__ tile_m,
               float* __restrict__ Yg,           // [PADROWS][Kout]
               int Ndim, int Kout) {
  const int e = tile_e[blockIdx.x];
  if (e < 0) return;
  const int row0 = tile_m[blockIdx.x];

  const int lane = threadIdx.x & 31;
  const int wid  = threadIdx.x >> 5;
  const int wm   = wid & 3;
  const int wn   = wid >> 2;
  const int m0   = row0 + wm * 32;
  const int j0   = blockIdx.y * 128 + wn * 64;
  const bf16_t* W = W2T + (size_t)e * Kout * Ndim;

  v8f acc[2][4];
  for (int i = 0; i < 2; ++i)
    for (int j = 0; j < 4; ++j) acc[i][j] = (v8f)(0.0f);

  for (int k = 0; k < Ndim; k += 32) {
    v16bf a0 = load_frag_A(Hg, m0,      k, Ndim, lane);
    v16bf a1 = load_frag_A(Hg, m0 + 16, k, Ndim, lane);
    v16bf b0 = load_frag_B(W, j0,      k, Ndim, lane);
    v16bf b1 = load_frag_B(W, j0 + 16, k, Ndim, lane);
    v16bf b2 = load_frag_B(W, j0 + 32, k, Ndim, lane);
    v16bf b3 = load_frag_B(W, j0 + 48, k, Ndim, lane);
    acc[0][0] = wmma_bf16(a0, b0, acc[0][0]);
    acc[0][1] = wmma_bf16(a0, b1, acc[0][1]);
    acc[0][2] = wmma_bf16(a0, b2, acc[0][2]);
    acc[0][3] = wmma_bf16(a0, b3, acc[0][3]);
    acc[1][0] = wmma_bf16(a1, b0, acc[1][0]);
    acc[1][1] = wmma_bf16(a1, b1, acc[1][1]);
    acc[1][2] = wmma_bf16(a1, b2, acc[1][2]);
    acc[1][3] = wmma_bf16(a1, b3, acc[1][3]);
  }

  const int nl = lane & 15;
  const int ml = (lane >> 4) << 3;
  for (int mi = 0; mi < 2; ++mi) {
    int pbase = m0 + mi * 16 + ml;
    for (int ni = 0; ni < 4; ++ni) {
      int j = j0 + ni * 16 + nl;
      for (int r = 0; r < 8; ++r)
        Yg[(long)(pbase + r) * Kout + j] = acc[mi][ni][r];
    }
  }
}

// ------------------------------ combine ------------------------------------
// out[m][j] = tw[m][0]*Yg[inv[2m]][j] + tw[m][1]*Yg[inv[2m+1]][j]
// Streaming float4 reduce, atomic-free, writes every output element.
__global__ void combine_out(const float* __restrict__ Yg,   // [PADROWS][K]
                            const int* __restrict__ inv,    // [NSLOT]
                            const float* __restrict__ tw,   // [M][TOPK]
                            float* __restrict__ Out, int Kdim) {
  int m  = blockIdx.x;
  int p0 = inv[2 * m], p1 = inv[2 * m + 1];
  float w0 = tw[2 * m], w1 = tw[2 * m + 1];
  const float* y0 = Yg + (long)p0 * Kdim;
  const float* y1 = Yg + (long)p1 * Kdim;
  float* o = Out + (long)m * Kdim;
  for (int j = threadIdx.x * 4; j < Kdim; j += blockDim.x * 4) {
    float4 a = *(const float4*)(y0 + j);
    float4 b = *(const float4*)(y1 + j);
    float4 r;
    r.x = w0 * a.x + w1 * b.x;
    r.y = w0 * a.y + w1 * b.y;
    r.z = w0 * a.z + w1 * b.z;
    r.w = w0 * a.w + w1 * b.w;
    *(float4*)(o + j) = r;
  }
}

// ------------------------------ launcher -----------------------------------

extern "C" void kernel_launch(void* const* d_in, const int* in_sizes, int n_in,
                              void* d_out, int out_size, void* d_ws, size_t ws_size,
                              hipStream_t stream) {
  (void)in_sizes; (void)n_in; (void)out_size; (void)ws_size;
  constexpr int M = MOE_M, K = MOE_K, E = MOE_E, N = MOE_N;

  const float*     hs = (const float*)d_in[0];        // [M][K]
  const float*     w1 = (const float*)d_in[1];        // [E][K][2N]
  const float*     w2 = (const float*)d_in[2];        // [E][N][K]
  const float*     tw = (const float*)d_in[3];        // [M][TOPK]
  const long long* ti = (const long long*)d_in[4];    // [M][TOPK] int64
  float*          out = (float*)d_out;                // [M][K]

  char*   ws     = (char*)d_ws;
  size_t  off    = 0;
  bf16_t* W1T    = (bf16_t*)(ws + off); off += (size_t)E * 2 * N * K * sizeof(bf16_t); // 32 MB
  bf16_t* W2T    = (bf16_t*)(ws + off); off += (size_t)E * K * N * sizeof(bf16_t);     // 16 MB
  bf16_t* Xg     = (bf16_t*)(ws + off); off += (size_t)PADROWS * K * sizeof(bf16_t);   // ~19 MB
  bf16_t* Hg     = (bf16_t*)(ws + off); off += (size_t)PADROWS * N * sizeof(bf16_t);   // ~19 MB
  float*  Yg     = (float*)(ws + off);  off += (size_t)PADROWS * K * sizeof(float);    // ~38 MB
  int*    perm   = (int*)(ws + off);    off += (size_t)PADROWS * sizeof(int);
  int*    inv    = (int*)(ws + off);    off += (size_t)NSLOT * sizeof(int);
  int*    tile_e = (int*)(ws + off);    off += (size_t)MAXTILES * sizeof(int);
  int*    tile_m = (int*)(ws + off);    off += (size_t)MAXTILES * sizeof(int);

  // prep: convert+transpose weights, build permutation, gather X
  transpose_cvt_bf16<<<4096, 256, 0, stream>>>(w1, W1T, E, K, 2 * N);
  transpose_cvt_bf16<<<4096, 256, 0, stream>>>(w2, W2T, E, N, K);
  build_perm<<<1, 256, 0, stream>>>(ti, perm, inv, tile_e, tile_m);
  gather_cvt_x<<<PADROWS, 256, 0, stream>>>(hs, perm, Xg, K);

  // grouped GEMMs over data-independent worst-case tile grids
  dim3 g1(MAXTILES, N / 64);    // 72 x 16
  dim3 g2(MAXTILES, K / 128);   // 72 x 8
  moe_gemm1_silu<<<g1, 256, 0, stream>>>(Xg, W1T, Hg, tile_e, tile_m, K, N);
  moe_gemm2<<<g2, 256, 0, stream>>>(Hg, W2T, tile_e, tile_m, Yg, N, K);

  // atomic-free weighted combine (writes every element of d_out)
  combine_out<<<M, 256, 0, stream>>>(Yg, inv, tw, out, K);
}
